// MambaPeft_5136780886283
// MI455X (gfx1250) — matrix operand order repaired and checked by reading, not compiled, not run on hardware
//
#include <hip/hip_runtime.h>
#include <hip/hip_bf16.h>

// ---------------- problem constants ----------------
#define D_MODEL  1024
#define D_STATE  16
#define D_CONV   4
#define D_INNER  2048
#define DT_RANK  64
#define BATCH    2
#define SEQLEN   1024
#define M_TOTAL  (BATCH * SEQLEN)   // 2048 rows in every GEMM
#define NX_PAD   128                // x_dbl width padded from 96 -> 128

typedef _Float16 half_t;
typedef __attribute__((ext_vector_type(16))) _Float16 v16h;
typedef __attribute__((ext_vector_type(8)))  _Float16 v8h;
typedef __attribute__((ext_vector_type(8)))  float    v8f;
typedef __attribute__((ext_vector_type(4)))  int      v4i;

#if defined(__gfx1250__) && __has_builtin(__builtin_amdgcn_global_load_async_to_lds_b128)
#define HAVE_ASYNC_LDS 1
#define GAS __attribute__((address_space(1)))
#define LAS __attribute__((address_space(3)))
#endif

__device__ __forceinline__ void wait_async_zero() {
#if defined(HAVE_ASYNC_LDS)
#if __has_builtin(__builtin_amdgcn_s_wait_asynccnt)
  __builtin_amdgcn_s_wait_asynccnt(0);
#else
  asm volatile("s_wait_asynccnt 0x0" ::: "memory");
#endif
#endif
}

// ---------------- f32 -> f16 convert (with zero pad) ----------------
__global__ void cvt_f32_f16(const float* __restrict__ src, half_t* __restrict__ dst,
                            int n_src, int n_dst) {
  int i = blockIdx.x * blockDim.x + threadIdx.x;
  if (i < n_dst) dst[i] = (i < n_src) ? (half_t)src[i] : (half_t)0.f;
}

// ---------------- extract dt_low (first 64 of padded 128 cols) ----------------
__global__ void extract_dtlow(const float* __restrict__ xdbl, half_t* __restrict__ dt) {
  int i = blockIdx.x * blockDim.x + threadIdx.x;   // M_TOTAL*64
  int r = i >> 6, c = i & 63;
  dt[i] = (half_t)xdbl[(size_t)r * NX_PAD + c];
}

// ---------------- WMMA f16 GEMM: C[M,N] = A[M,K] * W[N,K]^T ----------------
// N, K compile-time. MODE 0: plain f32 store. MODE 1: softplus(acc+bias[n]).
// Block: 128 threads (4 waves, 2x2). Wave tile 32x64 (2x4 WMMA tiles).
// Block tile: 64(M) x 128(N). K stepped by 32.
// B tile (128 rows x 32 K, f16) staged in LDS via async global->LDS copies,
// double-buffered; LDS row stride padded to 40 halfs for bank spread.
#define BSTRIDE 40   // halfs per LDS row (80B: 16B-aligned, conflict-free row map)

template <int MODE, int N, int K>
__global__ __launch_bounds__(128)
void gemm_f16(const half_t* __restrict__ A, const half_t* __restrict__ W,
              float* __restrict__ C, const float* __restrict__ bias) {
  static_assert(K % 32 == 0 && N % 128 == 0, "tile divisibility");
  __shared__ half_t Bs[2][128 * BSTRIDE];   // 2 x 10KB

  const int tid  = threadIdx.x;
  const int lane = tid & 31;
  const int wave = tid >> 5;
  const int row0 = blockIdx.y * 64 + (wave >> 1) * 32;
  const int colB = blockIdx.x * 128;        // block column base (global)
  const int colL = (wave & 1) * 64;         // wave column base (within block)

  // A fragment addressing (16-bit A 16x32; contiguous k = 16*(lane>>4) + s)
  const int am    = lane & 15;
  const int akoff = (lane >> 4) * 16;
  // B fragment addressing (16-bit B 32x16; contiguous k = 16*(lane>>4) + s)
  const int bn    = lane & 15;
  const int bkoff = (lane >> 4) * 16;

  // per-thread B staging: row (colB+tid), 32 halfs (64B) per k-step
  const half_t* wsrc = W + (size_t)(colB + tid) * K;
  half_t* ldst[2] = { &Bs[0][tid * BSTRIDE], &Bs[1][tid * BSTRIDE] };

  const v8f zero = {0.f, 0.f, 0.f, 0.f, 0.f, 0.f, 0.f, 0.f};
  v8f acc[2][4];
#pragma unroll
  for (int i = 0; i < 2; ++i)
#pragma unroll
    for (int j = 0; j < 4; ++j) acc[i][j] = zero;

  auto stage = [&](int k, int buf) {
    half_t* g = const_cast<half_t*>(wsrc + k);
    half_t* l = ldst[buf];
#if defined(HAVE_ASYNC_LDS)
    __builtin_amdgcn_global_load_async_to_lds_b128((GAS v4i*)g, (LAS v4i*)l, 0, 0);
    __builtin_amdgcn_global_load_async_to_lds_b128((GAS v4i*)g, (LAS v4i*)l, 16, 0);
    __builtin_amdgcn_global_load_async_to_lds_b128((GAS v4i*)g, (LAS v4i*)l, 32, 0);
    __builtin_amdgcn_global_load_async_to_lds_b128((GAS v4i*)g, (LAS v4i*)l, 48, 0);
#else
    *(v16h*)l        = *(const v16h*)g;
    *(v16h*)(l + 16) = *(const v16h*)(g + 16);
#endif
  };

  stage(0, 0);
  int buf = 0;
  for (int k = 0; k < K; k += 32, buf ^= 1) {
    wait_async_zero();      // this wave's async copies into Bs[buf] done
    __syncthreads();        // all waves' copies done; prior compute done
    if (k + 32 < K) stage(k + 32, buf ^ 1);   // overlap next copy with compute

    v16h a[2], b[4];
#pragma unroll
    for (int i = 0; i < 2; ++i) {
      const half_t* p = A + (size_t)(row0 + i * 16 + am) * K + k + akoff;
      v8h lo = *(const v8h*)(p);
    v8h hi = *(const v8h*)(p + 8);
      a[i] = __builtin_shufflevector(lo, hi, 0, 1, 2, 3, 4, 5, 6, 7,
                                             8, 9, 10, 11, 12, 13, 14, 15);
    }
#pragma unroll
    for (int j = 0; j < 4; ++j) {
      const half_t* q = &Bs[buf][(colL + j * 16 + bn) * BSTRIDE + bkoff];
      b[j] = *(const v16h*)(q);   // 2x ds_load_b128
    }
#pragma unroll
    for (int i = 0; i < 2; ++i)
#pragma unroll
      for (int j = 0; j < 4; ++j)
        acc[i][j] = __builtin_amdgcn_wmma_f32_16x16x32_f16(
            false, a[i], false, b[j], (short)0, acc[i][j], false, false);
  }

  // D layout: VGPR r -> row (lane>>4)*8 + r ; col = lane&15
  const int srow = (lane >> 4) * 8;
  const int scol = lane & 15;
#pragma unroll
  for (int i = 0; i < 2; ++i)
#pragma unroll
    for (int j = 0; j < 4; ++j) {
      const int cb = colB + colL + j * 16 + scol;
      float bval = (MODE == 1) ? bias[cb] : 0.f;
      float* cp = C + (size_t)(row0 + i * 16 + srow) * N + cb;
#pragma unroll
      for (int r = 0; r < 8; ++r) {     // immediate offsets r*N*4
        float v = acc[i][j][r];
        if (MODE == 1) {                // softplus(v + bias)
          v += bval;
          v = (v > 20.f) ? v : log1pf(__expf(v));
        }
        cp[(size_t)r * N] = v;
      }
    }
}

// ---------------- depthwise causal conv (k=4) + SiLU -> u (f16) ----------------
__global__ void conv_silu(const float* __restrict__ xz, const float* __restrict__ w,
                          const float* __restrict__ bias, half_t* __restrict__ u) {
  int d = blockIdx.x * blockDim.x + threadIdx.x;   // 0..2047
  int m = blockIdx.y;                              // 0..2047 (b*1024 + l)
  int l = m & (SEQLEN - 1);
  float acc = bias[d];
#pragma unroll
  for (int j = 0; j < D_CONV; ++j) {
    int t = l - (D_CONV - 1) + j;
    if (t >= 0)
      acc += xz[(size_t)(m - l + t) * (2 * D_INNER) + d] * w[d * D_CONV + j];
  }
  float s = acc / (1.f + __expf(-acc));            // SiLU
  u[(size_t)m * D_INNER + d] = (half_t)s;
}

// ---------------- selective scan: 16 lanes (states) per channel ----------------
__global__ __launch_bounds__(256)
void selective_scan(const float* __restrict__ delta, const half_t* __restrict__ u,
                    const float* __restrict__ xdbl, const float* __restrict__ A_log,
                    float* __restrict__ ys) {
  const int n  = threadIdx.x & 15;                      // state index
  const int ch = blockIdx.x * 16 + (threadIdx.x >> 4);  // 0..4095
  const int d  = ch & (D_INNER - 1);
  const int b  = ch >> 11;
  const int m0 = b * SEQLEN;

  const float Aval = -__expf(A_log[d * D_STATE + n]);
  float h = 0.f;
  for (int l = 0; l < SEQLEN; ++l) {
    const int m = m0 + l;
    const float dl = delta[(size_t)m * D_INNER + d];
    const float uu = (float)u[(size_t)m * D_INNER + d];
    const float Bv = xdbl[(size_t)m * NX_PAD + DT_RANK + n];
    const float Cv = xdbl[(size_t)m * NX_PAD + DT_RANK + D_STATE + n];
    const float dA = __expf(dl * Aval);
    h = fmaf(dA, h, dl * uu * Bv);
    float y = h * Cv;
    y += __shfl_xor(y, 1);     // 16-lane reduction (stays in half-wave)
    y += __shfl_xor(y, 2);
    y += __shfl_xor(y, 4);
    y += __shfl_xor(y, 8);
    if (n == 0) ys[(size_t)m * D_INNER + d] = y;
  }
}

// ---------------- gate: (ys + u*D) * silu(z) -> f16 ----------------
__global__ void gate_fuse(const float* __restrict__ ys, const half_t* __restrict__ u,
                          const float* __restrict__ xz, const float* __restrict__ Dp,
                          half_t* __restrict__ yb) {
  int d = blockIdx.x * blockDim.x + threadIdx.x;
  int m = blockIdx.y;
  float uu = (float)u[(size_t)m * D_INNER + d];
  float y  = ys[(size_t)m * D_INNER + d] + uu * Dp[d];
  float z  = xz[(size_t)m * (2 * D_INNER) + D_INNER + d];
  float g  = z / (1.f + __expf(-z));
  yb[(size_t)m * D_INNER + d] = (half_t)(y * g);
}

// ---------------- host launcher ----------------
extern "C" void kernel_launch(void* const* d_in, const int* in_sizes, int n_in,
                              void* d_out, int out_size, void* d_ws, size_t ws_size,
                              hipStream_t stream) {
  const float* hidden    = (const float*)d_in[0];
  const float* in_proj_w = (const float*)d_in[1];
  const float* conv_w    = (const float*)d_in[2];
  const float* conv_b    = (const float*)d_in[3];
  const float* x_proj_w  = (const float*)d_in[4];
  const float* dt_proj_w = (const float*)d_in[5];
  const float* dt_proj_b = (const float*)d_in[6];
  const float* A_log     = (const float*)d_in[7];
  const float* Dvec      = (const float*)d_in[8];
  const float* out_proj  = (const float*)d_in[9];
  float* out = (float*)d_out;

  // workspace carve-out (256B aligned)
  size_t off = 0;
  auto alloc = [&](size_t bytes) -> void* {
    void* p = (char*)d_ws + off;
    off += (bytes + 255) & ~(size_t)255;
    return p;
  };
  half_t* Hbf   = (half_t*)alloc((size_t)M_TOTAL * D_MODEL * 2);
  half_t* Winbf = (half_t*)alloc((size_t)2 * D_INNER * D_MODEL * 2);
  half_t* Xpbf  = (half_t*)alloc((size_t)NX_PAD * D_INNER * 2);
  half_t* Dtpbf = (half_t*)alloc((size_t)D_INNER * DT_RANK * 2);
  half_t* Opbf  = (half_t*)alloc((size_t)D_MODEL * D_INNER * 2);
  float*  xz    = (float*)alloc((size_t)M_TOTAL * 2 * D_INNER * 4);
  half_t* ubf   = (half_t*)alloc((size_t)M_TOTAL * D_INNER * 2);
  float*  xdbl  = (float*)alloc((size_t)M_TOTAL * NX_PAD * 4);
  half_t* dtlow = (half_t*)alloc((size_t)M_TOTAL * DT_RANK * 2);
  float*  delta = (float*)alloc((size_t)M_TOTAL * D_INNER * 4);
  float*  ys    = (float*)alloc((size_t)M_TOTAL * D_INNER * 4);
  half_t* ybf   = (half_t*)alloc((size_t)M_TOTAL * D_INNER * 2);
  (void)ws_size; (void)n_in; (void)in_sizes; (void)out_size;

  auto cvt = [&](const float* s, half_t* dst, int ns, int nd) {
    cvt_f32_f16<<<(nd + 255) / 256, 256, 0, stream>>>(s, dst, ns, nd);
  };
  cvt(hidden,    Hbf,   M_TOTAL * D_MODEL,      M_TOTAL * D_MODEL);
  cvt(in_proj_w, Winbf, 2 * D_INNER * D_MODEL,  2 * D_INNER * D_MODEL);
  cvt(x_proj_w,  Xpbf,  96 * D_INNER,           NX_PAD * D_INNER);   // zero-pad rows 96..127
  cvt(dt_proj_w, Dtpbf, D_INNER * DT_RANK,      D_INNER * DT_RANK);
  cvt(out_proj,  Opbf,  D_MODEL * D_INNER,      D_MODEL * D_INNER);

  // 1) xz = H @ Win^T   [2048 x 4096], K=1024
  gemm_f16<0, 2 * D_INNER, D_MODEL>
      <<<dim3((2 * D_INNER) / 128, M_TOTAL / 64), 128, 0, stream>>>(
      Hbf, Winbf, xz, nullptr);

  // 2) causal depthwise conv + SiLU -> u (f16)
  conv_silu<<<dim3(D_INNER / 256, M_TOTAL), 256, 0, stream>>>(xz, conv_w, conv_b, ubf);

  // 3) x_dbl = u @ Xp^T   [2048 x 128], K=2048
  gemm_f16<0, NX_PAD, D_INNER>
      <<<dim3(NX_PAD / 128, M_TOTAL / 64), 128, 0, stream>>>(
      ubf, Xpbf, xdbl, nullptr);

  // 4) dt_low (f16)
  extract_dtlow<<<(M_TOTAL * DT_RANK) / 256, 256, 0, stream>>>(xdbl, dtlow);

  // 5) delta = softplus(dt_low @ Dtp^T + b)   [2048 x 2048], K=64
  gemm_f16<1, D_INNER, DT_RANK>
      <<<dim3(D_INNER / 128, M_TOTAL / 64), 128, 0, stream>>>(
      dtlow, Dtpbf, delta, dt_proj_b);

  // 6) selective scan
  selective_scan<<<(BATCH * D_INNER) / 16, 256, 0, stream>>>(delta, ubf, xdbl, A_log, ys);

  // 7) gate
  gate_fuse<<<dim3(D_INNER / 256, M_TOTAL), 256, 0, stream>>>(ys, ubf, xz, Dvec, ybf);

  // 8) out = y @ Wout^T   [2048 x 1024], K=2048
  gemm_f16<0, D_MODEL, D_INNER>
      <<<dim3(D_MODEL / 128, M_TOTAL / 64), 128, 0, stream>>>(
      ybf, Opbf, out, nullptr);
}
